// OptimizedRA_66236985639269
// MI455X (gfx1250) — matrix operand order repaired and checked by reading, not compiled, not run on hardware
//
#include <hip/hip_runtime.h>
#include <hip/hip_bf16.h>

typedef __attribute__((ext_vector_type(16))) _Float16 v16h;
typedef __attribute__((ext_vector_type(8)))  float    v8f;
typedef unsigned int uint32x4 __attribute__((ext_vector_type(4)));
typedef int          int32x4  __attribute__((ext_vector_type(4)));
typedef int          int32x8  __attribute__((ext_vector_type(8)));

// Problem constants (match reference)
#define N_EMBD   1024
#define N_HEAD   16
#define HEAD_DIM 64
#define R_LOW    8
#define D_STD    56
#define BATCH    2
#define SEQ      2048
#define C3       (3 * N_EMBD)
#define BT       (BATCH * SEQ)          // 4096 rows
#define NEGBIG   (-1e9f)

#if defined(__HIP_DEVICE_COMPILE__) && __has_builtin(__builtin_amdgcn_tensor_load_to_lds)
#define HAVE_TDM 1
#else
#define HAVE_TDM 0
#endif

// ---------------------------------------------------------------------------
// WMMA fragment loaders (layouts per CDNA5 ISA 7.12.2, wave32)
// A 16x32 f16: lanes 0-15 row=base+lane,   chunks K=[k0..k0+7],[k0+16..k0+23]
//              lanes16-31 row=base+lane-16, chunks K=[k0+8..15],[k0+24..31]
// ---------------------------------------------------------------------------
__device__ inline v16h load_frag_a(const _Float16* p, int ld, int row, int k0, int lane) {
    int r    = lane & 15;
    int koff = k0 + ((lane >> 4) << 3);
    const _Float16* base = p + (size_t)(row + r) * ld + koff;
    v16h out;
    ((uint4*)&out)[0] = *(const uint4*)(base);
    ((uint4*)&out)[1] = *(const uint4*)(base + 16);
    return out;
}

// B 32x16 f16 (computing A·W^T with W stored (N,K) row-major):
// lanes 0-15: N=col+lane, K=[k0..k0+7],[k0+16..k0+23]; lanes16-31: N=col+lane-16, K=[k0+8..k0+15],[k0+24..k0+31]
__device__ inline v16h load_frag_b(const _Float16* p, int ld, int col, int k0, int lane) {
    int n    = lane & 15;
    int koff = k0 + ((lane >> 4) << 3);
    const _Float16* base = p + (size_t)(col + n) * ld + koff;
    v16h out;
    ((uint4*)&out)[0] = *(const uint4*)(base);
    ((uint4*)&out)[1] = *(const uint4*)(base + 16);
    return out;
}

// ---------------------------------------------------------------------------
// TDM: stage a 64-row x 32-f16 B tile (row stride K elements) into LDS.
// D# layout per CDNA5 ISA 08_async_tensor §8.3/8.4:
//  group0: [1:0]=count=1, [63:32]=lds_addr, [120:64]=global_addr, [127:126]=2
//  group1: [17:16]=data_size(1=2B), [79:48]=tensor_dim0, [111:80]=tensor_dim1,
//          [127:112]=tile_dim0, [143:128]=tile_dim1, [207:160]=dim0_stride
// ---------------------------------------------------------------------------
__device__ inline void tdm_issue_b_tile(const _Float16* gsrc, int ldK, _Float16* ldsDst) {
#if HAVE_TDM
    unsigned long long ga = (unsigned long long)(size_t)gsrc;
    unsigned int lds = (unsigned int)(size_t)ldsDst;   // flat LDS addr: [31:0] = LDS offset
    uint32x4 g0 = {};
    g0[0] = 1u;                                        // count=1 (valid user descriptor)
    g0[1] = lds;                                       // lds_addr
    g0[2] = (unsigned int)ga;                          // global_addr[31:0]
    g0[3] = ((unsigned int)(ga >> 32) & 0x01FFFFFFu) | (2u << 30);  // addr[56:32] | type=2
    int32x8 g1 = {};
    g1[0] = (int)(1u << 16);          // data_size = 1 (2 bytes)
    g1[1] = (int)(32u << 16);         // tensor_dim0[15:0] = 32
    g1[2] = (int)(64u << 16);         // tensor_dim0 hi = 0 | tensor_dim1[15:0] = 64
    g1[3] = (int)(32u << 16);         // tensor_dim1 hi = 0 | tile_dim0 = 32
    g1[4] = 64;                       // tile_dim1 = 64 (tile_dim2 = 0)
    g1[5] = ldK;                      // tensor_dim0_stride[31:0] = K elements
    g1[6] = 0;
    g1[7] = 0;
    int32x4 z4 = {};
#if __clang_major__ >= 23
    int32x8 z8 = {};
    __builtin_amdgcn_tensor_load_to_lds(g0, g1, z4, z4, z8, 0);
#else
    __builtin_amdgcn_tensor_load_to_lds(g0, g1, z4, z4, 0);
#endif
#else
    (void)gsrc; (void)ldK; (void)ldsDst;
#endif
}

// Cooperative fallback (also used when TDM builtin is unavailable)
__device__ inline void stage_b_tile(const _Float16* gsrc, int ldK, _Float16* ldsDst, int warp) {
#if HAVE_TDM
    if (warp == 0) tdm_issue_b_tile(gsrc, ldK, ldsDst);
#else
    for (int i = threadIdx.x; i < 64 * 4; i += 128) {   // 64 rows x 4 uint4 per row
        int row = i >> 2, seg = i & 3;
        ((uint4*)(ldsDst + row * 32))[seg] = *(const uint4*)(gsrc + (size_t)row * ldK + seg * 8);
    }
#endif
}

__device__ inline void stage_wait(int warp) {
#if HAVE_TDM
    if (warp == 0) __builtin_amdgcn_s_wait_tensorcnt(0);
#else
    (void)warp;
#endif
}

// ---------------------------------------------------------------------------
// Kernel: fp32 -> f16 conversion
// ---------------------------------------------------------------------------
__global__ void cvt_f32_f16(const float* __restrict__ src, _Float16* __restrict__ dst, int n) {
    int i = blockIdx.x * blockDim.x + threadIdx.x;
    if (i < n) dst[i] = (_Float16)src[i];
}

// ---------------------------------------------------------------------------
// Kernel: C(M,N) fp32 = A(M,K) f16 · B(N,K)^T f16   via v_wmma_f32_16x16x32_f16
// block = 128 threads (4 waves); block tile = 128(M) x 64(N).
// wave w -> rows [bx*128 + w*32, +32) as 2 M-tiles, all 4 N-tiles.
// B tile (64 x 32, shared by all waves) staged into LDS by the Tensor Data
// Mover, double-buffered: TDM fill of buffer j+1 overlaps WMMAs on buffer j.
// ---------------------------------------------------------------------------
__global__ __launch_bounds__(128) void wmma_gemm_nt(const _Float16* __restrict__ A,
                                                    const _Float16* __restrict__ B,
                                                    float* __restrict__ C,
                                                    int M, int N, int K) {
    __shared__ __align__(16) _Float16 Bs[2][64 * 32];

    const int lane    = threadIdx.x & 31;
    const int warp    = threadIdx.x >> 5;
    const int rowBase = blockIdx.x * 128 + warp * 32;
    const int colBase = blockIdx.y * 64;

    v8f acc0[4] = {}, acc1[4] = {};

    // prologue: stage first B tile
    stage_b_tile(B + (size_t)colBase * K, K, &Bs[0][0], warp);

    int buf = 0;
    for (int k0 = 0; k0 < K; k0 += 32) {
        stage_wait(warp);        // TDM for current buffer complete (issuer wave)
        __syncthreads();         // publish buffer to all waves

        // kick off next tile fill (runs on TDM engine during our WMMAs)
        if (k0 + 32 < K)
            stage_b_tile(B + (size_t)colBase * K + (k0 + 32), K, &Bs[buf ^ 1][0], warp);

        if (k0 + 64 < K)
            __builtin_prefetch(A + (size_t)(rowBase + (lane & 15)) * K + k0 + 64, 0, 3);

        v16h a0 = load_frag_a(A, K, rowBase,      k0, lane);
        v16h a1 = load_frag_a(A, K, rowBase + 16, k0, lane);
        const _Float16* Bc = &Bs[buf][0];
#pragma unroll
        for (int j = 0; j < 4; ++j) {
            v16h bf = load_frag_b(Bc, 32, 16 * j, 0, lane);   // ds_load_b128 x2
            acc0[j] = __builtin_amdgcn_wmma_f32_16x16x32_f16(
                false, a0, false, bf, (short)0, acc0[j], false, false);
            acc1[j] = __builtin_amdgcn_wmma_f32_16x16x32_f16(
                false, a1, false, bf, (short)0, acc1[j], false, false);
        }
        __syncthreads();         // all reads of buf done before it is refilled
        buf ^= 1;
    }

    const int n    = lane & 15;
    const int rOff = (lane >> 4) << 3;
#pragma unroll
    for (int j = 0; j < 4; ++j)
#pragma unroll
        for (int r = 0; r < 8; ++r) {
            C[(size_t)(rowBase + rOff + r) * N + colBase + 16 * j + n]      = acc0[j][r];
            C[(size_t)(rowBase + 16 + rOff + r) * N + colBase + 16 * j + n] = acc1[j][r];
        }
}

// ---------------------------------------------------------------------------
// Kernel: build q_aug / k_aug (f16, (B,H,T,64), scale folded into q_aug) and
// V^T (f16, (B,H,64,T)) from fp32 qkv (B,T,3C). One thread per (b,t,h).
// ---------------------------------------------------------------------------
__global__ void prep_aug(const float* __restrict__ qkv,
                         const float* __restrict__ W_recip,
                         const float* __restrict__ w_std,
                         const float* __restrict__ w_rec,
                         _Float16* __restrict__ q_aug,
                         _Float16* __restrict__ k_aug,
                         _Float16* __restrict__ v_t) {
    int idx = blockIdx.x * blockDim.x + threadIdx.x;
    if (idx >= BATCH * SEQ * N_HEAD) return;
    int h = idx % N_HEAD;
    int t = (idx / N_HEAD) % SEQ;
    int b = idx / (N_HEAD * SEQ);

    const float* qp = qkv + (size_t)(b * SEQ + t) * C3 + h * HEAD_DIM;
    const float* kp = qp + N_EMBD;
    const float* vp = qp + 2 * N_EMBD;

    float q[HEAD_DIM], k[HEAD_DIM];
#pragma unroll
    for (int d = 0; d < HEAD_DIM; ++d) { q[d] = qp[d]; k[d] = kp[d]; }

    float ql[R_LOW] = {}, kl[R_LOW] = {};
    for (int d = 0; d < HEAD_DIM; ++d) {
        float wq = q[d], wk = k[d];
#pragma unroll
        for (int r = 0; r < R_LOW; ++r) {
            float w = W_recip[d * R_LOW + r];
            ql[r] += wq * w;
            kl[r] += wk * w;
        }
    }
    float ss = sqrtf(w_std[h]);
    float sr = sqrtf(w_rec[h]);
    const float scl = 0.125f;  // 1/sqrt(HEAD_DIM), folded into q_aug

    size_t base = ((size_t)(b * N_HEAD + h) * SEQ + t) * HEAD_DIM;
#pragma unroll
    for (int d = 0; d < D_STD; ++d) {
        q_aug[base + d] = (_Float16)(scl * ss * q[d]);
        k_aug[base + d] = (_Float16)(ss * k[d]);
    }
#pragma unroll
    for (int r = 0; r < R_LOW; ++r) {
        // cross augmentation: q side carries k_low, k side carries q_low
        q_aug[base + D_STD + r] = (_Float16)(scl * sr * kl[r]);
        k_aug[base + D_STD + r] = (_Float16)(sr * ql[r]);
    }
    size_t vbase = (size_t)(b * N_HEAD + h) * HEAD_DIM * SEQ;
#pragma unroll
    for (int d = 0; d < HEAD_DIM; ++d)
        v_t[vbase + (size_t)d * SEQ + t] = (_Float16)vp[d];
}

// ---------------------------------------------------------------------------
// Kernel: causal flash attention with per-head key bias.
// block = 128 (4 waves); each wave owns one 16-query tile of one (b,h).
// Per 32-key block: 4 QK^T WMMAs -> bias/mask -> online softmax -> P via LDS
// (layout change C->A, s_wait_dscnt) -> 4 PV WMMAs. EXEC stays all-ones.
// ---------------------------------------------------------------------------
__global__ __launch_bounds__(128) void flash_attn(const _Float16* __restrict__ q_aug,
                                                  const _Float16* __restrict__ k_aug,
                                                  const _Float16* __restrict__ v_t,
                                                  const float* __restrict__ w_disc,
                                                  const float* __restrict__ d_bias,
                                                  _Float16* __restrict__ attn_out) {
    __shared__ __align__(16) _Float16 Plds[4][16 * 32];

    const int lane = threadIdx.x & 31;
    const int warp = threadIdx.x >> 5;
    const int gw   = blockIdx.x * 4 + warp;          // 4096 waves total
    const int qt   = gw & 127;                        // SEQ/16 = 128 q-tiles
    const int h    = (gw >> 7) & 15;
    const int b    = gw >> 11;
    const int qbase = qt * 16;

    const _Float16* Q  = q_aug + (size_t)(b * N_HEAD + h) * SEQ * HEAD_DIM;
    const _Float16* Km = k_aug + (size_t)(b * N_HEAD + h) * SEQ * HEAD_DIM;
    const _Float16* Vt = v_t   + (size_t)(b * N_HEAD + h) * HEAD_DIM * SEQ;

    // Q fragments (K=64 -> two 16x32 A fragments), loaded once
    v16h aq0 = load_frag_a(Q, HEAD_DIM, qbase, 0, lane);
    v16h aq1 = load_frag_a(Q, HEAD_DIM, qbase, 32, lane);

    float m8[8], l8[8];
#pragma unroll
    for (int r = 0; r < 8; ++r) { m8[r] = -3.0e38f; l8[r] = 0.0f; }
    v8f acco[4] = {};

    const float wd   = w_disc[h];
    const int   n0l  = lane & 15;
    const int   rOff = (lane >> 4) << 3;
    _Float16*   Pw   = &Plds[warp][0];

    for (int k0 = 0; k0 < qbase + 16; k0 += 32) {
        // ---- scores: two 16-key tiles, K-dim 64 in two steps ----
        v8f s0 = {}, s1 = {};
        v16h b00 = load_frag_b(Km, HEAD_DIM, k0,      0,  lane);
        v16h b01 = load_frag_b(Km, HEAD_DIM, k0,      32, lane);
        v16h b10 = load_frag_b(Km, HEAD_DIM, k0 + 16, 0,  lane);
        v16h b11 = load_frag_b(Km, HEAD_DIM, k0 + 16, 32, lane);
        s0 = __builtin_amdgcn_wmma_f32_16x16x32_f16(false, aq0, false, b00, (short)0, s0, false, false);
        s0 = __builtin_amdgcn_wmma_f32_16x16x32_f16(false, aq1, false, b01, (short)0, s0, false, false);
        s1 = __builtin_amdgcn_wmma_f32_16x16x32_f16(false, aq0, false, b10, (short)0, s1, false, false);
        s1 = __builtin_amdgcn_wmma_f32_16x16x32_f16(false, aq1, false, b11, (short)0, s1, false, false);

        const int key0 = k0 + n0l;
        const int key1 = key0 + 16;
        const float bias0 = wd * d_bias[h * SEQ + key0];
        const float bias1 = wd * d_bias[h * SEQ + key1];

        // ---- bias + causal mask + online softmax (per C-layout row) ----
#pragma unroll
        for (int r = 0; r < 8; ++r) {
            const int row = qbase + rOff + r;
            float v0 = s0[r] + bias0; if (key0 > row) v0 = NEGBIG;
            float v1 = s1[r] + bias1; if (key1 > row) v1 = NEGBIG;
            float mx = fmaxf(v0, v1);
#pragma unroll
            for (int off = 8; off >= 1; off >>= 1)
                mx = fmaxf(mx, __shfl_xor(mx, off, 32));   // stays within 16-lane half
            const float mn = fmaxf(m8[r], mx);
            const float sc = __expf(m8[r] - mn);
            const float p0 = __expf(v0 - mn);
            const float p1 = __expf(v1 - mn);
            float rs = p0 + p1;
#pragma unroll
            for (int off = 8; off >= 1; off >>= 1)
                rs += __shfl_xor(rs, off, 32);
            l8[r] = l8[r] * sc + rs;
            m8[r] = mn;
            acco[0][r] *= sc; acco[1][r] *= sc;
            acco[2][r] *= sc; acco[3][r] *= sc;
            // stage P (C-layout -> LDS row-major 16x32)
            Pw[(rOff + r) * 32 + n0l]      = (_Float16)p0;
            Pw[(rOff + r) * 32 + n0l + 16] = (_Float16)p1;
        }
        // wave-private LDS: drain DS writes before cross-lane reads
        asm volatile("s_wait_dscnt 0" ::: "memory");

        // ---- P·V: reload P in A layout, V^T columns as B operand ----
        v16h pf = load_frag_a(Pw, 32, 0, 0, lane);
#pragma unroll
        for (int j = 0; j < 4; ++j) {
            v16h vf = load_frag_b(Vt, SEQ, 16 * j, k0, lane);
            acco[j] = __builtin_amdgcn_wmma_f32_16x16x32_f16(
                false, pf, false, vf, (short)0, acco[j], false, false);
        }
    }

    // ---- epilogue: normalize and scatter into (B,T,C) f16 for projection ----
    const size_t obase = (size_t)b * SEQ * N_EMBD;
#pragma unroll
    for (int j = 0; j < 4; ++j) {
        const int d = h * HEAD_DIM + 16 * j + n0l;
#pragma unroll
        for (int r = 0; r < 8; ++r) {
            const int row = qbase + rOff + r;
            attn_out[obase + (size_t)row * N_EMBD + d] = (_Float16)(acco[j][r] / l8[r]);
        }
    }
}

// ---------------------------------------------------------------------------
// Host-side orchestration
// ---------------------------------------------------------------------------
static inline size_t align256(size_t x) { return (x + 255) & ~(size_t)255; }

extern "C" void kernel_launch(void* const* d_in, const int* in_sizes, int n_in,
                              void* d_out, int out_size, void* d_ws, size_t ws_size,
                              hipStream_t stream) {
    const float* x       = (const float*)d_in[0];   // (B,T,C)
    const float* Wqkv    = (const float*)d_in[1];   // (3C,C)
    const float* Wproj   = (const float*)d_in[2];   // (C,C)
    const float* W_recip = (const float*)d_in[3];   // (64,8)
    const float* w_std   = (const float*)d_in[4];   // (H)
    const float* w_rec   = (const float*)d_in[5];   // (H)
    const float* w_disc  = (const float*)d_in[6];   // (H)
    const float* d_bias  = (const float*)d_in[7];   // (H,2048)
    float*       out     = (float*)d_out;           // (B,T,C) fp32

    // workspace partition
    char* ws = (char*)d_ws;
    size_t off = 0;
    _Float16* x_h    = (_Float16*)(ws + off); off = align256(off + (size_t)BT * N_EMBD * 2);
    _Float16* Wqkv_h = (_Float16*)(ws + off); off = align256(off + (size_t)C3 * N_EMBD * 2);
    _Float16* Wproj_h= (_Float16*)(ws + off); off = align256(off + (size_t)N_EMBD * N_EMBD * 2);
    float*    qkv    = (float*)   (ws + off); off = align256(off + (size_t)BT * C3 * 4);
    _Float16* q_aug  = (_Float16*)(ws + off); off = align256(off + (size_t)BT * N_EMBD * 2);
    _Float16* k_aug  = (_Float16*)(ws + off); off = align256(off + (size_t)BT * N_EMBD * 2);
    _Float16* v_t    = (_Float16*)(ws + off); off = align256(off + (size_t)BT * N_EMBD * 2);
    _Float16* a_out  = (_Float16*)(ws + off); off = align256(off + (size_t)BT * N_EMBD * 2);
    (void)ws_size; (void)in_sizes; (void)n_in; (void)out_size;

    // 1) fp32 -> f16 conversions
    {
        int n0 = BT * N_EMBD, n1 = C3 * N_EMBD, n2 = N_EMBD * N_EMBD;
        cvt_f32_f16<<<(n0 + 255) / 256, 256, 0, stream>>>(x, x_h, n0);
        cvt_f32_f16<<<(n1 + 255) / 256, 256, 0, stream>>>(Wqkv, Wqkv_h, n1);
        cvt_f32_f16<<<(n2 + 255) / 256, 256, 0, stream>>>(Wproj, Wproj_h, n2);
    }
    // 2) qkv = x @ Wqkv^T   (4096 x 3072 x 1024), TDM-staged B tiles
    wmma_gemm_nt<<<dim3(BT / 128, C3 / 64), 128, 0, stream>>>(x_h, Wqkv_h, qkv, BT, C3, N_EMBD);

    // 3) augmentation + V transpose
    {
        int n = BATCH * SEQ * N_HEAD;
        prep_aug<<<(n + 255) / 256, 256, 0, stream>>>(qkv, W_recip, w_std, w_rec, q_aug, k_aug, v_t);
    }
    // 4) causal flash attention (4096 q-tile waves / 4 waves per block)
    flash_attn<<<1024, 128, 0, stream>>>(q_aug, k_aug, v_t, w_disc, d_bias, a_out);

    // 5) out = attn_out @ Wproj^T  (4096 x 1024 x 1024), fp32 result
    wmma_gemm_nt<<<dim3(BT / 128, N_EMBD / 64), 128, 0, stream>>>(a_out, Wproj_h, out, BT, N_EMBD, N_EMBD);
}
